// SpeechTopKDecoder_4672924418414
// MI455X (gfx1250) — matrix operand order repaired and checked by reading, not compiled, not run on hardware
//
#include <hip/hip_runtime.h>

// ============================ problem dims =================================
#define NEGF (-1.0e9f)
constexpr int cB = 16, cK = 3, cS = 512, cH = 1024, cC = 2000, cT = 64;
constexpr int cNH = 4, cDH = 256, cL = 2;
constexpr int BK  = cB * cK;      // 48 beam rows
constexpr int G4H = 4 * cH;       // 4096 gate width
constexpr int SOS = 1, EOS = 2;

// persistent decoder launch shape
constexpr int NB = 48, TPB = 256, NT = NB * TPB, NW = NT / 32;  // 384 waves

// ============================ WMMA types ===================================
typedef __attribute__((ext_vector_type(16))) __bf16 v16bf;
typedef __attribute__((ext_vector_type(8)))  float  v8f;

// ======================= workspace layout (bytes) ==========================
constexpr size_t OFF_BAR   = 0;                            // 2 x u32 barrier
constexpr size_t OFF_STATE = 256;                          // scA,scB,inp,pred
constexpr size_t OFF_KH    = 1024;
constexpr size_t SZ_KH     = (size_t)cB * cS * cH * 2;     // bf16 K cache
constexpr size_t OFF_VH    = OFF_KH  + SZ_KH;
constexpr size_t OFF_EMB   = OFF_VH  + SZ_KH;
constexpr size_t SZ_EMB    = (size_t)cC * cH * 2;
constexpr size_t OFF_WIH   = OFF_EMB + SZ_EMB;
constexpr size_t SZ_WIH    = (size_t)cL * G4H * cH * 2;
constexpr size_t OFF_WHH   = OFF_WIH + SZ_WIH;
constexpr size_t OFF_WQT   = OFF_WHH + SZ_WIH;
constexpr size_t SZ_HH     = (size_t)cH * cH * 2;
constexpr size_t OFF_WPT   = OFF_WQT + SZ_HH;
constexpr size_t OFF_WGT   = OFF_WPT + SZ_HH;
constexpr size_t SZ_WGT    = (size_t)cC * cH * 2;
constexpr size_t OFF_WKT   = OFF_WGT + SZ_WGT;
constexpr size_t OFF_WVT   = OFF_WKT + SZ_HH;
constexpr size_t OFF_ENC   = OFF_WVT + SZ_HH;              // bf16 encoder (precompute only)
constexpr size_t SZ_ENC    = (size_t)cB * cS * cH * 2;
// ---- decode-time scratch aliases the dead enc-bf16 region ----
constexpr size_t SZ_ROWBF  = (size_t)BK * cH * 2;          // 48x1024 bf16
constexpr size_t OFF_XBF   = OFF_ENC;
constexpr size_t OFF_H0P   = OFF_XBF + SZ_ROWBF;
constexpr size_t OFF_H1P   = OFF_H0P + SZ_ROWBF;
constexpr size_t OFF_H0N   = OFF_H1P + SZ_ROWBF;
constexpr size_t OFF_H1N   = OFF_H0N + SZ_ROWBF;
constexpr size_t OFF_O1B   = OFF_H1N + SZ_ROWBF;
constexpr size_t OFF_THB   = OFF_O1B + SZ_ROWBF;
constexpr size_t SZ_HBUF   = (size_t)2 * cL * BK * cH * 4; // double-buffered
constexpr size_t OFF_H     = OFF_THB + SZ_ROWBF;
constexpr size_t OFF_C     = OFF_H + SZ_HBUF;
constexpr size_t OFF_GATE  = OFF_C + SZ_HBUF;
constexpr size_t SZ_GATE   = (size_t)BK * G4H * 4;
constexpr size_t SZ_ROWF   = (size_t)BK * cH * 4;
constexpr size_t OFF_Q     = OFF_GATE + SZ_GATE;
constexpr size_t OFF_CTX   = OFF_Q   + SZ_ROWF;
constexpr size_t OFF_O1    = OFF_CTX + SZ_ROWF;
constexpr size_t OFF_T2    = OFF_O1  + SZ_ROWF;
constexpr size_t OFF_ATT   = OFF_T2  + SZ_ROWF;            // 48*4*512 f32

// ============================ helpers ======================================
__device__ inline unsigned short f2bf(float f) {
  unsigned u = __float_as_uint(f);
  return (unsigned short)((u + 0x7fffu + ((u >> 16) & 1u)) >> 16);   // RNE
}
__device__ inline float bf2f(unsigned short s) {
  return __uint_as_float(((unsigned)s) << 16);
}
__device__ inline float sigm(float x) { return 1.f / (1.f + expf(-x)); }

__device__ inline float wred_max(float v) {
  #pragma unroll
  for (int o = 16; o; o >>= 1) v = fmaxf(v, __shfl_xor(v, o, 32));
  return v;
}
__device__ inline float wred_sum(float v) {
  #pragma unroll
  for (int o = 16; o; o >>= 1) v += __shfl_xor(v, o, 32);
  return v;
}

// A fragment: 16x32 bf16 A tile in this code's contiguous k order (not the ISA table order):
// lane(lo) = row, half = lane>>4 -> K in {half*16..half*16+15}
__device__ inline v16bf ldA(const unsigned short* p) {
  union { uint4 q[2]; v16bf v; } u;
  u.q[0] = *(const uint4*)(p);
  u.q[1] = *(const uint4*)(p + 8);
  return u.v;
}
// B fragment: lanes 0-15 hold K=0..15, lanes 16-31 hold K=16..31 of col=lane&15
__device__ inline v16bf ldB(const unsigned short* p) {
  union { uint4 q[2]; v16bf v; } u;
  u.q[0] = *(const uint4*)(p);
  u.q[1] = *(const uint4*)(p + 8);
  return u.v;
}
__device__ inline v8f bfwmma(v16bf a, v16bf b, v8f c) {
  return __builtin_amdgcn_wmma_f32_16x16x32_bf16(false, a, false, b,
                                                 (short)0, c, false, false);
}

// out[m][n] += sum_k A[m][k] * Wt[n][k], all bf16, ld = K-dim stride
__device__ inline v8f tile_gemm(const unsigned short* A, const unsigned short* W,
                                int lda, int ldw, int m0, int n0, int K, v8f acc) {
  const int lane = threadIdx.x & 31, half = lane >> 4, lo = lane & 15;
  const unsigned short* pa = A + (size_t)(m0 + lo) * lda + half * 16;
  const unsigned short* pb = W + (size_t)(n0 + lo) * ldw + half * 16;
  #pragma unroll 4
  for (int k0 = 0; k0 < K; k0 += 32) {
    __builtin_prefetch(pb + k0 + 256, 0, 1);   // global_prefetch_b8
    acc = bfwmma(ldA(pa + k0), ldB(pb + k0), acc);
  }
  return acc;
}
// dual-A version: x@Wi^T + h@Wh^T accumulated into one tile
__device__ inline v8f tile_gemm2(const unsigned short* A0, const unsigned short* W0,
                                 const unsigned short* A1, const unsigned short* W1,
                                 int m0, int n0, int K, v8f acc) {
  const int lane = threadIdx.x & 31, half = lane >> 4, lo = lane & 15;
  const unsigned short* pa0 = A0 + (size_t)(m0 + lo) * cH + half * 16;
  const unsigned short* pb0 = W0 + (size_t)(n0 + lo) * cH + half * 16;
  const unsigned short* pa1 = A1 + (size_t)(m0 + lo) * cH + half * 16;
  const unsigned short* pb1 = W1 + (size_t)(n0 + lo) * cH + half * 16;
  #pragma unroll 2
  for (int k0 = 0; k0 < K; k0 += 32) {
    __builtin_prefetch(pb0 + k0 + 256, 0, 1);
    acc = bfwmma(ldA(pa0 + k0), ldB(pb0 + k0), acc);
    acc = bfwmma(ldA(pa1 + k0), ldB(pb1 + k0), acc);
  }
  return acc;
}

// device-wide generational barrier (all NB blocks resident)
__device__ inline void gsync(unsigned* bar, unsigned nb) {
  __syncthreads();
  if (threadIdx.x == 0) {
    __threadfence();
    volatile unsigned* vgen = bar + 1;
    unsigned g = *vgen;
    if (atomicAdd(bar, 1u) == nb - 1u) {
      bar[0] = 0u;
      __threadfence();
      atomicAdd(bar + 1, 1u);
    } else {
      while (*vgen == g) { __builtin_amdgcn_s_sleep(1); }
    }
    __threadfence();
  }
  __syncthreads();
}

// ========================= conversion kernels ==============================
__global__ void conv_plain(const float* __restrict__ src,
                           unsigned short* __restrict__ dst, size_t n) {
  size_t stride = (size_t)gridDim.x * blockDim.x;
  for (size_t i = (size_t)blockIdx.x * blockDim.x + threadIdx.x; i < n; i += stride)
    dst[i] = f2bf(src[i]);
}
// src (R x C) row-major -> dst[c*R + r]  (so weight rows become contiguous-k)
__global__ void conv_T(const float* __restrict__ src,
                       unsigned short* __restrict__ dst, int R, int C) {
  size_t n = (size_t)R * C, stride = (size_t)gridDim.x * blockDim.x;
  for (size_t i = (size_t)blockIdx.x * blockDim.x + threadIdx.x; i < n; i += stride) {
    int r = (int)(i / C), c = (int)(i % C);
    dst[(size_t)c * R + r] = f2bf(src[i]);
  }
}

// ======================= K/V cache precompute ==============================
// kh[b*S+s][n] = bf16( enc[b,s,:] . wkT[n,:] + bk[n] ),  same for vh.
__global__ void __launch_bounds__(256) kv_kernel(char* ws, const float* __restrict__ bkb,
                                                 const float* __restrict__ bvb) {
  const unsigned short* encb = (const unsigned short*)(ws + OFF_ENC);
  const unsigned short* wkt  = (const unsigned short*)(ws + OFF_WKT);
  const unsigned short* wvt  = (const unsigned short*)(ws + OFF_WVT);
  unsigned short* kh = (unsigned short*)(ws + OFF_KH);
  unsigned short* vh = (unsigned short*)(ws + OFF_VH);
  const int gid = blockIdx.x * blockDim.x + threadIdx.x;
  const int wid = gid >> 5, lane = threadIdx.x & 31;
  const int wstride = (gridDim.x * blockDim.x) >> 5;
  const int half = lane >> 4, lo = lane & 15;
  // 2 matrices x (512 x 64) 16x16 tiles over M=8192,N=1024,K=1024
  for (int tIdx = wid; tIdx < 2 * 512 * 64; tIdx += wstride) {
    int mat = tIdx >> 15, rest = tIdx & 32767;
    int m0 = (rest >> 6) * 16, n0 = (rest & 63) * 16;
    const unsigned short* W = mat ? wvt : wkt;
    const float* bias = mat ? bvb : bkb;
    unsigned short* D = mat ? vh : kh;
    v8f acc = {};
    acc = tile_gemm(encb, W, cH, cH, m0, n0, cH, acc);
    int n = n0 + lo;
    float bb = bias[n];
    #pragma unroll
    for (int r = 0; r < 8; r++)
      D[(size_t)(m0 + r + half * 8) * cH + n] = f2bf(acc[r] + bb);
  }
}

// ====================== persistent beam-search decoder =====================
__global__ void __launch_bounds__(TPB, 1)
decode_kernel(char* ws,
              const float* __restrict__ b_ih, const float* __restrict__ b_hh,
              const float* __restrict__ bq,   const float* __restrict__ bp,
              const float* __restrict__ ln1g, const float* __restrict__ ln1b,
              const float* __restrict__ ln2g, const float* __restrict__ ln2b,
              float* __restrict__ out) {
  unsigned* bar = (unsigned*)(ws + OFF_BAR);
  float* scA = (float*)(ws + OFF_STATE);
  float* scB = scA + BK;
  int*   inp  = (int*)(scA + 2 * BK);
  int*   pred = inp + BK;

  const unsigned short* emb = (const unsigned short*)(ws + OFF_EMB);
  const unsigned short* wih = (const unsigned short*)(ws + OFF_WIH);
  const unsigned short* whh = (const unsigned short*)(ws + OFF_WHH);
  const unsigned short* wqt = (const unsigned short*)(ws + OFF_WQT);
  const unsigned short* wpt = (const unsigned short*)(ws + OFF_WPT);
  const unsigned short* wgt = (const unsigned short*)(ws + OFF_WGT);
  const unsigned short* kh  = (const unsigned short*)(ws + OFF_KH);
  const unsigned short* vh  = (const unsigned short*)(ws + OFF_VH);

  unsigned short* xbf  = (unsigned short*)(ws + OFF_XBF);
  unsigned short* h0p  = (unsigned short*)(ws + OFF_H0P);
  unsigned short* h1p  = (unsigned short*)(ws + OFF_H1P);
  unsigned short* h0n  = (unsigned short*)(ws + OFF_H0N);
  unsigned short* h1n  = (unsigned short*)(ws + OFF_H1N);
  unsigned short* o1b  = (unsigned short*)(ws + OFF_O1B);
  unsigned short* thb  = (unsigned short*)(ws + OFF_THB);
  float* hb    = (float*)(ws + OFF_H);
  float* cbuf  = (float*)(ws + OFF_C);
  float* gates = (float*)(ws + OFF_GATE);
  float* qb    = (float*)(ws + OFF_Q);
  float* ctx   = (float*)(ws + OFF_CTX);
  float* o1f   = (float*)(ws + OFF_O1);
  float* t2f   = (float*)(ws + OFF_T2);
  float* att   = (float*)(ws + OFF_ATT);

  const int tid = threadIdx.x;
  const int gid = blockIdx.x * TPB + tid;
  const int lane = tid & 31;
  const int wid = gid >> 5;                 // 0..383
  constexpr int HBUF = cL * BK * cH;        // floats per h/c buffer
  constexpr int RH = BK * cH;               // 49152

  // ---- init h,c = 0, scores, SOS inputs ----
  for (int i = gid; i < 2 * HBUF; i += NT) { hb[i] = 0.f; cbuf[i] = 0.f; }
  for (int i = gid; i < BK; i += NT) { scA[i] = (i % cK == 0) ? 0.f : NEGF; inp[i] = SOS; }
  gsync(bar, NB);

  #pragma unroll 1
  for (int t = 0; t < cT; t++) {
    float* hcur = hb   + (t & 1) * HBUF;
    float* hnxt = hb   + ((t + 1) & 1) * HBUF;
    float* ccur = cbuf + (t & 1) * HBUF;
    float* cnxt = cbuf + ((t + 1) & 1) * HBUF;
    float* scc  = (t & 1) ? scB : scA;
    float* scn  = (t & 1) ? scA : scB;
    float* lrow0 = out + (size_t)t * BK * cC;

    // P1: embedding lookup + bf16 of gathered h0/h1
    for (int i = gid; i < RH; i += NT) {
      int m = i >> 10, k = i & (cH - 1);
      xbf[i] = emb[(size_t)inp[m] * cH + k];
      h0p[i] = f2bf(hcur[i]);
      h1p[i] = f2bf(hcur[RH + i]);
    }
    gsync(bar, NB);

    // P2..P5: two LSTM layers (WMMA gate GEMM + activations)
    #pragma unroll 1
    for (int l = 0; l < cL; l++) {
      const unsigned short* Abf = (l == 0) ? xbf : h0n;
      const unsigned short* Hbf = (l == 0) ? h0p : h1p;
      const unsigned short* Wi = wih + (size_t)l * G4H * cH;
      const unsigned short* Wh = whh + (size_t)l * G4H * cH;
      const float* bi = b_ih + l * G4H;
      const float* bh = b_hh + l * G4H;
      for (int tI = wid; tI < 3 * 256; tI += NW) {       // 48x4096 tiles
        int m0 = (tI >> 8) * 16, n0 = (tI & 255) * 16;
        v8f acc = {};
        acc = tile_gemm2(Abf, Wi, Hbf, Wh, m0, n0, cH, acc);
        int half = lane >> 4, lo = lane & 15, n = n0 + lo;
        float bias = bi[n] + bh[n];
        #pragma unroll
        for (int r = 0; r < 8; r++)
          gates[(size_t)(m0 + r + half * 8) * G4H + n] = acc[r] + bias;
      }
      gsync(bar, NB);
      unsigned short* hout = (l == 0) ? h0n : h1n;
      for (int i = gid; i < RH; i += NT) {
        int m = i >> 10, n = i & (cH - 1);
        const float* gr = gates + (size_t)m * G4H;
        float ig = sigm(gr[n]), fg = sigm(gr[cH + n]);
        float gg = tanhf(gr[2 * cH + n]), og = sigm(gr[3 * cH + n]);
        float cl = fg * ccur[l * RH + i] + ig * gg;
        float hl = og * tanhf(cl);
        ccur[l * RH + i] = cl;
        hcur[l * RH + i] = hl;
        hout[i] = f2bf(hl);
      }
      gsync(bar, NB);
    }

    // P6: q = x @ wq + bq   (x = new h1)
    for (int tI = wid; tI < 3 * 64; tI += NW) {
      int m0 = (tI >> 6) * 16, n0 = (tI & 63) * 16;
      v8f acc = {};
      acc = tile_gemm(h1n, wqt, cH, cH, m0, n0, cH, acc);
      int half = lane >> 4, lo = lane & 15, n = n0 + lo;
      float bb = bq[n];
      #pragma unroll
      for (int r = 0; r < 8; r++)
        qb[(size_t)(m0 + r + half * 8) * cH + n] = acc[r] + bb;
    }
    gsync(bar, NB);

    // P7: attention scores  att[(m,hh,s)] = q.k / sqrt(DH)
    for (int iT = gid; iT < BK * cNH * cS; iT += NT) {
      int s = iT & (cS - 1), rh = iT >> 9;
      int m = rh >> 2, hh = rh & 3;
      const float* qp = qb + (size_t)m * cH + hh * cDH;
      const unsigned short* kp = kh + ((size_t)(m / cK) * cS + s) * cH + hh * cDH;
      float acc = 0.f;
      for (int d = 0; d < cDH; d += 8) {
        uint4 kv4 = *(const uint4*)(kp + d);
        const unsigned short* k8 = (const unsigned short*)&kv4;
        #pragma unroll
        for (int e = 0; e < 8; e++) acc += qp[d + e] * bf2f(k8[e]);
      }
      att[(size_t)rh * cS + s] = acc * 0.0625f;   // 1/sqrt(256)
    }
    gsync(bar, NB);

    // P8: softmax over S, one wave per (m,hh)
    if (wid < BK * cNH) {
      float* ap = att + (size_t)wid * cS;
      float mx = NEGF;
      for (int s = lane; s < cS; s += 32) mx = fmaxf(mx, ap[s]);
      mx = wred_max(mx);
      float sum = 0.f;
      for (int s = lane; s < cS; s += 32) { float e = expf(ap[s] - mx); ap[s] = e; sum += e; }
      sum = wred_sum(sum);
      float inv = 1.f / sum;
      for (int s = lane; s < cS; s += 32) ap[s] *= inv;
    }
    gsync(bar, NB);

    // P9: ctx = att @ v
    for (int iT = gid; iT < RH; iT += NT) {
      int m = iT >> 10, n = iT & (cH - 1), hh = n >> 8;
      const float* ap = att + (size_t)(m * cNH + hh) * cS;
      const unsigned short* vp = vh + (size_t)(m / cK) * cS * cH + n;
      float acc = 0.f;
      for (int s = 0; s < cS; s++) acc += ap[s] * bf2f(vp[(size_t)s * cH]);
      ctx[iT] = acc;
    }
    gsync(bar, NB);

    // P10: out1 = LN(ctx + x), one wave per row
    if (wid < BK) {
      const float* xr = hcur + RH + (size_t)wid * cH;
      const float* cr = ctx + (size_t)wid * cH;
      float s = 0.f;
      for (int n = lane; n < cH; n += 32) s += cr[n] + xr[n];
      float mean = wred_sum(s) * (1.f / cH);
      float v = 0.f;
      for (int n = lane; n < cH; n += 32) { float d = cr[n] + xr[n] - mean; v += d * d; }
      float rs = rsqrtf(wred_sum(v) * (1.f / cH) + 1e-5f);
      float* orow = o1f + (size_t)wid * cH;
      unsigned short* obf = o1b + (size_t)wid * cH;
      for (int n = lane; n < cH; n += 32) {
        float o = (cr[n] + xr[n] - mean) * rs * ln1g[n] + ln1b[n];
        orow[n] = o; obf[n] = f2bf(o);
      }
    }
    gsync(bar, NB);

    // P11: t2 = out1 @ wp + bp + out1 (residual)
    for (int tI = wid; tI < 3 * 64; tI += NW) {
      int m0 = (tI >> 6) * 16, n0 = (tI & 63) * 16;
      v8f acc = {};
      acc = tile_gemm(o1b, wpt, cH, cH, m0, n0, cH, acc);
      int half = lane >> 4, lo = lane & 15, n = n0 + lo;
      float bb = bp[n];
      #pragma unroll
      for (int r = 0; r < 8; r++) {
        int m = m0 + r + half * 8;
        t2f[(size_t)m * cH + n] = acc[r] + bb + o1f[(size_t)m * cH + n];
      }
    }
    gsync(bar, NB);

    // P12: out2 = LN(t2); th = tanh(out2)
    if (wid < BK) {
      const float* tr = t2f + (size_t)wid * cH;
      float s = 0.f;
      for (int n = lane; n < cH; n += 32) s += tr[n];
      float mean = wred_sum(s) * (1.f / cH);
      float v = 0.f;
      for (int n = lane; n < cH; n += 32) { float d = tr[n] - mean; v += d * d; }
      float rs = rsqrtf(wred_sum(v) * (1.f / cH) + 1e-5f);
      unsigned short* tb = thb + (size_t)wid * cH;
      for (int n = lane; n < cH; n += 32) {
        float o = (tr[n] - mean) * rs * ln2g[n] + ln2b[n];
        tb[n] = f2bf(tanhf(o));
      }
    }
    gsync(bar, NB);

    // P13: logits = th @ wg  (48 x 2000), written straight to d_out slice
    for (int tI = wid; tI < 3 * 125; tI += NW) {
      int m0 = (tI / 125) * 16, n0 = (tI % 125) * 16;
      v8f acc = {};
      acc = tile_gemm(thb, wgt, cH, cH, m0, n0, cH, acc);
      int half = lane >> 4, lo = lane & 15, n = n0 + lo;
      #pragma unroll
      for (int r = 0; r < 8; r++)
        lrow0[(size_t)(m0 + r + half * 8) * cC + n] = acc[r];
    }
    gsync(bar, NB);

    // P14: in-place log_softmax per row
    if (wid < BK) {
      float* lr = lrow0 + (size_t)wid * cC;
      float mx = NEGF;
      for (int n = lane; n < cC; n += 32) mx = fmaxf(mx, lr[n]);
      mx = wred_max(mx);
      float sum = 0.f;
      for (int n = lane; n < cC; n += 32) sum += expf(lr[n] - mx);
      float lse = logf(wred_sum(sum)) + mx;
      for (int n = lane; n < cC; n += 32) lr[n] -= lse;
    }
    gsync(bar, NB);

    // P15: top-3 over cand[b][kk*C+c] = sc + logp, one wave per batch
    if (wid < cB) {
      int b = wid;
      int e0 = -1, e1 = -1;
      #pragma unroll 1
      for (int kk = 0; kk < cK; kk++) {
        float best = NEGF * 4.f; int bidx = 0x7fffffff;
        for (int j = lane; j < cK * cC; j += 32) {
          if (j == e0 || j == e1) continue;
          int r = b * cK + j / cC;
          float val = scc[r] + lrow0[(size_t)r * cC + (j % cC)];
          if (val > best || (val == best && j < bidx)) { best = val; bidx = j; }
        }
        #pragma unroll
        for (int o = 16; o; o >>= 1) {
          float ov = __shfl_xor(best, o, 32);
          int   oi = __shfl_xor(bidx, o, 32);
          if (ov > best || (ov == best && oi < bidx)) { best = ov; bidx = oi; }
        }
        if (lane == 0) {
          int r = b * cK + kk;
          int sym = bidx % cC;
          inp[r]  = sym;
          pred[r] = bidx / cC + b * cK;
          scn[r]  = (sym == EOS) ? NEGF : best;
        }
        if (kk == 0) e0 = bidx; else if (kk == 1) e1 = bidx;
      }
    }
    gsync(bar, NB);

    // P16: beam gather of h,c into next buffers
    for (int i = gid; i < cL * RH; i += NT) {
      int l = i / RH, rem = i - l * RH, m = rem >> 10, k = rem & (cH - 1);
      int p = pred[m];
      hnxt[i] = hcur[l * RH + p * cH + k];
      cnxt[i] = ccur[l * RH + p * cH + k];
    }
    gsync(bar, NB);
  }
}

// ============================== host side ==================================
extern "C" void kernel_launch(void* const* d_in, const int* in_sizes, int n_in,
                              void* d_out, int out_size, void* d_ws, size_t ws_size,
                              hipStream_t stream) {
  (void)in_sizes; (void)n_in; (void)out_size; (void)ws_size;
  char* ws = (char*)d_ws;
  const float* enc   = (const float*)d_in[0];
  const float* embf  = (const float*)d_in[1];
  const float* w_ih  = (const float*)d_in[2];
  const float* w_hh  = (const float*)d_in[3];
  const float* b_ih  = (const float*)d_in[4];
  const float* b_hh  = (const float*)d_in[5];
  const float* wq    = (const float*)d_in[6];
  const float* bqv   = (const float*)d_in[7];
  const float* wk    = (const float*)d_in[8];
  const float* bkv   = (const float*)d_in[9];
  const float* wv    = (const float*)d_in[10];
  const float* bvv   = (const float*)d_in[11];
  const float* ln1g  = (const float*)d_in[12];
  const float* ln1b  = (const float*)d_in[13];
  const float* wp    = (const float*)d_in[14];
  const float* bpv   = (const float*)d_in[15];
  const float* ln2g  = (const float*)d_in[16];
  const float* ln2b  = (const float*)d_in[17];
  const float* wg    = (const float*)d_in[18];

  hipMemsetAsync(ws + OFF_BAR, 0, 1024, stream);   // barrier + state region

  // fp32 -> bf16 conversions (plus weight transposes to contiguous-k form)
  conv_plain<<<1024, 256, 0, stream>>>(embf, (unsigned short*)(ws + OFF_EMB),
                                       (size_t)cC * cH);
  conv_plain<<<2048, 256, 0, stream>>>(w_ih, (unsigned short*)(ws + OFF_WIH),
                                       (size_t)cL * G4H * cH);
  conv_plain<<<2048, 256, 0, stream>>>(w_hh, (unsigned short*)(ws + OFF_WHH),
                                       (size_t)cL * G4H * cH);
  conv_plain<<<2048, 256, 0, stream>>>(enc,  (unsigned short*)(ws + OFF_ENC),
                                       (size_t)cB * cS * cH);
  conv_T<<<1024, 256, 0, stream>>>(wq, (unsigned short*)(ws + OFF_WQT), cH, cH);
  conv_T<<<1024, 256, 0, stream>>>(wk, (unsigned short*)(ws + OFF_WKT), cH, cH);
  conv_T<<<1024, 256, 0, stream>>>(wv, (unsigned short*)(ws + OFF_WVT), cH, cH);
  conv_T<<<1024, 256, 0, stream>>>(wp, (unsigned short*)(ws + OFF_WPT), cH, cH);
  conv_T<<<1024, 256, 0, stream>>>(wg, (unsigned short*)(ws + OFF_WGT), cH, cC);

  // K/V cache precompute (deduplicated across beams)
  kv_kernel<<<2048, 256, 0, stream>>>(ws, bkv, bvv);

  // persistent 64-step beam-search decode
  decode_kernel<<<NB, TPB, 0, stream>>>(ws, b_ih, b_hh, bqv, bpv,
                                        ln1g, ln1b, ln2g, ln2b, (float*)d_out);
}
